// EncoderLayer_57698590654771
// MI455X (gfx1250) — compile-verified, matrix operand order checked by reading, not run on hardware
//
#include <hip/hip_runtime.h>
#include <hip/hip_bf16.h>

// ---- CDNA5 (gfx1250) wave32 WMMA transformer encoder layer ----
// B=16, T=1024, D=1024, H=16, HD=64, DFF=4096
// GEMMs: double-buffered LDS pipeline fed by GLOBAL_LOAD_ASYNC_TO_LDS_B128
// (ASYNCcnt), K-step 64 => 8 v_wmma_f32_16x16x32_f16 per barrier per wave.
// Flash-style all-WMMA attention with online softmax.

typedef _Float16 half_t;
typedef __attribute__((ext_vector_type(16))) _Float16 v16h;
typedef __attribute__((ext_vector_type(8)))  float    v8f;

union Acc8 { v8f v; float f[8]; };

static __device__ __forceinline__ v8f wmma_f32_f16(v16h a, v16h b, v8f c) {
  // D = A(16x32 f16) * B(32x16 f16) + C(16x16 f32)
  return __builtin_amdgcn_wmma_f32_16x16x32_f16(
      /*neg_a=*/false, a, /*neg_b=*/false, b,
      /*c_mod=*/(short)0, c, /*reuse_a=*/false, /*reuse_b=*/false);
}

// CDNA5 async copy: LDS[vdst_off] = MEM[gaddr], 16 bytes per lane, ASYNCcnt.
// Generic shared pointers carry the in-allocation LDS byte offset in their
// low 32 bits (flat aperture mapping, ISA 10.2) == the VDST operand.
static __device__ __forceinline__ void async_copy_b128(void* lds, const half_t* gaddr) {
  unsigned loff = (unsigned)(uintptr_t)lds;
  asm volatile("global_load_async_to_lds_b128 %0, %1, off"
               :: "v"(loff), "v"(gaddr) : "memory");
}
static __device__ __forceinline__ void wait_async0() {
  asm volatile("s_wait_asynccnt 0" ::: "memory");
}

// Load a 16x32 f16 fragment (A-layout; B-layout identical with N for M) from
// a row-major source: element (r, k) at p[r*ld + k]. ISA 7.12.2.
static __device__ __forceinline__ v16h load_frag_rm(const half_t* p, int ld, int lane) {
  union { v16h v; unsigned u[8]; } f;
  const int r = lane & 15, g = lane >> 4;
  const half_t* row = p + (size_t)r * ld + 8 * g;
  const unsigned* q0 = (const unsigned*)row;          // K = 8g .. 8g+7
  const unsigned* q1 = (const unsigned*)(row + 16);   // K = 16+8g ..
#pragma unroll
  for (int i = 0; i < 4; ++i) { f.u[i] = q0[i]; f.u[4 + i] = q1[i]; }
  return f.v;
}

// Same fragment gathered from a K-major source: element (k, n) at p[k*ld+n].
static __device__ __forceinline__ v16h load_frag_cm(const half_t* p, int ld, int lane) {
  union { v16h v; unsigned u[8]; } f;
  const int n = lane & 15, g = lane >> 4;
#pragma unroll
  for (int i = 0; i < 8; ++i) {
    const int kb = ((i < 4) ? 2 * i : 16 + 2 * (i - 4)) + 8 * g;
    unsigned lo = *(const unsigned short*)(p + (size_t)kb * ld + n);
    unsigned hi = *(const unsigned short*)(p + (size_t)(kb + 1) * ld + n);
    f.u[i] = lo | (hi << 16);
  }
  return f.v;
}

static __device__ __forceinline__ float rowmax16(float x) {
#pragma unroll
  for (int o = 1; o < 16; o <<= 1) x = fmaxf(x, __shfl_xor(x, o, 32));
  return x;
}
static __device__ __forceinline__ float rowsum16(float x) {
#pragma unroll
  for (int o = 1; o < 16; o <<= 1) x += __shfl_xor(x, o, 32);
  return x;
}

// LDS tile geometry: K-step 64 halves per stage; rows padded to 72 halves
// (144B = 16B-aligned for async b128 stores; 36-bank row skew keeps the 16
// rows a lane-group reads on distinct banks: 36*r mod 64 is a permutation).
#define APAD  72
#define KSTEP 64

// Double-buffered block GEMM body: block tile 128(M) x 64(N), 8 waves, wave
// tile 32x32 (2x2 frags), 8 WMMAs per barrier. Tiles staged via async copy.
static __device__ __forceinline__ void wmma_block_gemm(
    const half_t* __restrict__ A, const half_t* __restrict__ Bt, int K,
    int rowBlk, int colBlk, int tid, int lane, int w,
    half_t Ash[2][128][APAD], half_t Bsh[2][64][APAD], Acc8 acc[2][2]) {
#pragma unroll
  for (int i = 0; i < 2; ++i)
#pragma unroll
    for (int j = 0; j < 2; ++j)
#pragma unroll
      for (int r = 0; r < 8; ++r) acc[i][j].f[r] = 0.f;

  const int wr = (w >> 1) * 32;  // wave row offset in tile
  const int wc = (w & 1) * 32;   // wave col offset in tile

  // Copy slices: A tile = 128 rows x 128B -> 1024 b128 chunks (4/thread);
  //              B tile =  64 rows x 128B ->  512 chunks (2/thread).
  auto issue_tile = [&](int buf, int kb) {
#pragma unroll
    for (int j = 0; j < 4; ++j) {
      const int id = tid + 256 * j, row = id >> 3, c = (id & 7) * 8;
      async_copy_b128(&Ash[buf][row][c], A + (size_t)(rowBlk + row) * K + kb + c);
    }
#pragma unroll
    for (int j = 0; j < 2; ++j) {
      const int id = tid + 256 * j, row = id >> 3, c = (id & 7) * 8;
      async_copy_b128(&Bsh[buf][row][c], Bt + (size_t)(colBlk + row) * K + kb + c);
    }
  };

  int buf = 0;
  issue_tile(0, 0);
  wait_async0();
  __syncthreads();

  for (int kb = 0; kb < K; kb += KSTEP) {
    if (kb + KSTEP < K) issue_tile(buf ^ 1, kb + KSTEP);
    if (kb + 2 * KSTEP < K) {  // keep the HBM stream warm: global_prefetch_b8
      __builtin_prefetch(A + (size_t)(rowBlk + (tid >> 3)) * K + kb + 2 * KSTEP, 0, 1);
      __builtin_prefetch(Bt + (size_t)(colBlk + (tid >> 3 & 63)) * K + kb + 2 * KSTEP, 0, 1);
    }
#pragma unroll
    for (int ks = 0; ks < KSTEP; ks += 32) {
      v16h a0 = load_frag_rm(&Ash[buf][wr][ks], APAD, lane);
      v16h a1 = load_frag_rm(&Ash[buf][wr + 16][ks], APAD, lane);
      v16h b0 = load_frag_rm(&Bsh[buf][wc][ks], APAD, lane);
      v16h b1 = load_frag_rm(&Bsh[buf][wc + 16][ks], APAD, lane);
      acc[0][0].v = wmma_f32_f16(a0, b0, acc[0][0].v);
      acc[0][1].v = wmma_f32_f16(a0, b1, acc[0][1].v);
      acc[1][0].v = wmma_f32_f16(a1, b0, acc[1][0].v);
      acc[1][1].v = wmma_f32_f16(a1, b1, acc[1][1].v);
    }
    wait_async0();     // my async writes into buf^1 have landed in LDS
    __syncthreads();   // everyone done reading buf and writing buf^1
    buf ^= 1;
  }
}

// ---------------- weight convert + transpose: Wt[n][k] = (f16)W[k][n] -------
__global__ void convert_transpose_kernel(const float* __restrict__ W,
                                         half_t* __restrict__ Wt, int K, int N) {
  size_t idx = (size_t)blockIdx.x * blockDim.x + threadIdx.x;
  if (idx >= (size_t)K * N) return;
  int k = (int)(idx / N), n = (int)(idx % N);
  Wt[(size_t)n * K + k] = (half_t)W[idx];
}

// ---------------- LayerNorm over D=1024, one block (256 thr) per row --------
__global__ __launch_bounds__(256)
void ln_kernel(const float* __restrict__ X, const float* __restrict__ g,
               const float* __restrict__ bb, half_t* __restrict__ outH,
               float* __restrict__ outF) {
  const int row = blockIdx.x, tid = threadIdx.x;
  const float* xr = X + (size_t)row * 1024;
  float v[4], s = 0.f;
#pragma unroll
  for (int i = 0; i < 4; ++i) { v[i] = xr[tid + 256 * i]; s += v[i]; }
  __shared__ float red[256];
  red[tid] = s; __syncthreads();
  for (int o = 128; o > 0; o >>= 1) { if (tid < o) red[tid] += red[tid + o]; __syncthreads(); }
  const float mu = red[0] * (1.f / 1024.f);
  __syncthreads();
  s = 0.f;
#pragma unroll
  for (int i = 0; i < 4; ++i) { float d = v[i] - mu; s += d * d; }
  red[tid] = s; __syncthreads();
  for (int o = 128; o > 0; o >>= 1) { if (tid < o) red[tid] += red[tid + o]; __syncthreads(); }
  const float rs = rsqrtf(red[0] * (1.f / 1024.f) + 1e-5f);
#pragma unroll
  for (int i = 0; i < 4; ++i) {
    int c = tid + 256 * i;
    float y = (v[i] - mu) * rs * g[c] + bb[c];
    if (outH) outH[(size_t)row * 1024 + c] = (half_t)y;
    if (outF) outF[(size_t)row * 1024 + c] = y;
  }
}

// ---------------- generic WMMA GEMM: C = A(f16,MxK) * Bt(f16,NxK)^T ---------
__global__ __launch_bounds__(256)
void gemm_kernel(const half_t* __restrict__ A, const half_t* __restrict__ Bt,
                 const float* __restrict__ bias, const float* __restrict__ resid,
                 float* __restrict__ outF, half_t* __restrict__ outH,
                 int M, int N, int K, int relu) {
  const int tid = threadIdx.x, lane = tid & 31, w = tid >> 5;
  const int rowBlk = blockIdx.y * 128, colBlk = blockIdx.x * 64;
  __shared__ half_t Ash[2][128][APAD];
  __shared__ half_t Bsh[2][64][APAD];
  Acc8 acc[2][2];
  wmma_block_gemm(A, Bt, K, rowBlk, colBlk, tid, lane, w, Ash, Bsh, acc);

  const int rowBase = rowBlk + (w >> 1) * 32, colBase = colBlk + (w & 1) * 32;
#pragma unroll
  for (int mi = 0; mi < 2; ++mi)
#pragma unroll
    for (int ni = 0; ni < 2; ++ni)
#pragma unroll
      for (int r = 0; r < 8; ++r) {
        const int row = rowBase + mi * 16 + r + 8 * (lane >> 4);
        const int col = colBase + ni * 16 + (lane & 15);
        float val = acc[mi][ni].f[r] + bias[col];
        if (resid) val += resid[(size_t)row * N + col];
        if (relu) val = fmaxf(val, 0.f);
        if (outF) outF[(size_t)row * N + col] = val;
        if (outH) outH[(size_t)row * N + col] = (half_t)val;
      }
}

// ---------------- QKV GEMM (M=16384, N=3072, K=1024) with head scatter ------
__global__ __launch_bounds__(256)
void gemm_qkv_kernel(const half_t* __restrict__ A, const half_t* __restrict__ Bt,
                     const float* __restrict__ bias, half_t* __restrict__ q,
                     half_t* __restrict__ k, half_t* __restrict__ v) {
  const int tid = threadIdx.x, lane = tid & 31, w = tid >> 5;
  const int rowBlk = blockIdx.y * 128, colBlk = blockIdx.x * 64;
  __shared__ half_t Ash[2][128][APAD];
  __shared__ half_t Bsh[2][64][APAD];
  Acc8 acc[2][2];
  wmma_block_gemm(A, Bt, 1024, rowBlk, colBlk, tid, lane, w, Ash, Bsh, acc);

  const int rowBase = rowBlk + (w >> 1) * 32, colBase = colBlk + (w & 1) * 32;
#pragma unroll
  for (int mi = 0; mi < 2; ++mi)
#pragma unroll
    for (int ni = 0; ni < 2; ++ni)
#pragma unroll
      for (int r = 0; r < 8; ++r) {
        const int row = rowBase + mi * 16 + r + 8 * (lane >> 4);  // b*T + t
        const int col = colBase + ni * 16 + (lane & 15);          // 0..3071
        const float val = acc[mi][ni].f[r] + bias[col];
        const int head = col / 192, rem = col - head * 192;
        const int which = rem >> 6, hd = rem & 63;
        const int b = row >> 10, t = row & 1023;
        const size_t dst = ((size_t)(b * 16 + head) * 1024 + t) * 64 + hd;
        half_t* dp = (which == 0) ? q : (which == 1) ? k : v;
        dp[dst] = (half_t)val;
      }
}

// ---------------- flash-style attention, all-WMMA, wave32 -------------------
// grid = B*H*(T/128); 8 waves/block, each wave owns 16 query rows.
__global__ __launch_bounds__(256)
void attn_kernel(const half_t* __restrict__ Q, const half_t* __restrict__ Kx,
                 const half_t* __restrict__ V, half_t* __restrict__ O) {
  const int lane = threadIdx.x & 31, w = threadIdx.x >> 5;
  const int bh = blockIdx.x >> 3, qblk = blockIdx.x & 7;
  const int b = bh >> 4, h = bh & 15;
  const int qt0 = qblk * 128 + w * 16;
  const size_t headBase = (size_t)bh * 1024 * 64;
  __shared__ half_t Pst[8][16][32];  // per-wave private P staging (8KB)

  v16h qA0 = load_frag_rm(Q + headBase + (size_t)qt0 * 64, 64, lane);
  v16h qA1 = load_frag_rm(Q + headBase + (size_t)qt0 * 64 + 32, 64, lane);
  {  // fold 1/sqrt(HD)=0.125 into Q
    union { v16h v; _Float16 hh[16]; } u0, u1;
    u0.v = qA0; u1.v = qA1;
#pragma unroll
    for (int i = 0; i < 16; ++i) { u0.hh[i] *= (_Float16)0.125f; u1.hh[i] *= (_Float16)0.125f; }
    qA0 = u0.v; qA1 = u1.v;
  }

  float m[8], l[8];
  Acc8 oacc[4];
#pragma unroll
  for (int r = 0; r < 8; ++r) { m[r] = -1e30f; l[r] = 0.f; }
#pragma unroll
  for (int f = 0; f < 4; ++f)
#pragma unroll
    for (int r = 0; r < 8; ++r) oacc[f].f[r] = 0.f;

  for (int kt = 0; kt < 1024; kt += 32) {
    const half_t* kb = Kx + headBase + (size_t)kt * 64;
    Acc8 s0, s1;
#pragma unroll
    for (int r = 0; r < 8; ++r) { s0.f[r] = 0.f; s1.f[r] = 0.f; }
    // S(16q x 32k) = Q(16x64) * K(32x64)^T : B operand = key rows, A-style
    s0.v = wmma_f32_f16(qA0, load_frag_rm(kb, 64, lane), s0.v);
    s0.v = wmma_f32_f16(qA1, load_frag_rm(kb + 32, 64, lane), s0.v);
    s1.v = wmma_f32_f16(qA0, load_frag_rm(kb + 16 * 64, 64, lane), s1.v);
    s1.v = wmma_f32_f16(qA1, load_frag_rm(kb + 16 * 64 + 32, 64, lane), s1.v);

    // online softmax per accumulator row (rows striped across 16-lane halves)
#pragma unroll
    for (int r = 0; r < 8; ++r) {
      const float a = s0.f[r], c = s1.f[r];
      const float mx = rowmax16(fmaxf(a, c));
      const float mn = fmaxf(m[r], mx);
      const float corr = __expf(m[r] - mn);
      const float p0 = __expf(a - mn), p1 = __expf(c - mn);
      const float rsum = rowsum16(p0 + p1);
      m[r] = mn;
      l[r] = l[r] * corr + rsum;
#pragma unroll
      for (int f = 0; f < 4; ++f) oacc[f].f[r] *= corr;
      const int prow = r + 8 * (lane >> 4);
      Pst[w][prow][lane & 15] = (half_t)p0;
      Pst[w][prow][16 + (lane & 15)] = (half_t)p1;
    }
    asm volatile("s_wait_dscnt 0" ::: "memory");  // P visible before re-read

    // O(16x64) += P(16x32) * V(32x64); V gathered key-major
    v16h pA = load_frag_rm(&Pst[w][0][0], 32, lane);
    const half_t* vb = V + headBase + (size_t)kt * 64;
#pragma unroll
    for (int f = 0; f < 4; ++f)
      oacc[f].v = wmma_f32_f16(pA, load_frag_cm(vb + f * 16, 64, lane), oacc[f].v);
  }

#pragma unroll
  for (int r = 0; r < 8; ++r) {
    const float inv = 1.f / l[r];
    const int t = qt0 + r + 8 * (lane >> 4);
    const size_t base = ((size_t)(b * 1024 + t)) * 1024 + h * 64 + (lane & 15);
#pragma unroll
    for (int f = 0; f < 4; ++f)
      O[base + f * 16] = (half_t)(oacc[f].f[r] * inv);
  }
}

// ---------------------------------------------------------------------------
extern "C" void kernel_launch(void* const* d_in, const int* in_sizes, int n_in,
                              void* d_out, int out_size, void* d_ws, size_t ws_size,
                              hipStream_t stream) {
  const float* x    = (const float*)d_in[0];
  const float* Wqkv = (const float*)d_in[1];
  const float* bqkv = (const float*)d_in[2];
  const float* Wo   = (const float*)d_in[3];
  const float* bo   = (const float*)d_in[4];
  const float* W1   = (const float*)d_in[5];
  const float* b1   = (const float*)d_in[6];
  const float* W2   = (const float*)d_in[7];
  const float* b2   = (const float*)d_in[8];
  const float* g1   = (const float*)d_in[9];
  const float* be1  = (const float*)d_in[10];
  const float* g2   = (const float*)d_in[11];
  const float* be2  = (const float*)d_in[12];

  const size_t BT = 16384;
  char* p = (char*)d_ws;
  auto alloc = [&](size_t bytes) { void* r = (void*)p; p += (bytes + 255) & ~(size_t)255; return r; };

  half_t* h16    = (half_t*)alloc(BT * 1024 * 2);        // LN1 out (dead after QKV)
  half_t* q16    = (half_t*)alloc(BT * 1024 * 2);        // dead after attention
  half_t* k16    = (half_t*)alloc(BT * 1024 * 2);
  half_t* v16    = (half_t*)alloc(BT * 1024 * 2);
  float*  r32    = (float*)alloc(BT * 1024 * 4);
  float*  r232   = (float*)alloc(BT * 1024 * 4);
  half_t* f116   = (half_t*)alloc(BT * 4096 * 2);
  half_t* wqkvT  = (half_t*)alloc((size_t)3072 * 1024 * 2);
  half_t* woT    = (half_t*)alloc((size_t)1024 * 1024 * 2);
  half_t* w1T    = (half_t*)alloc((size_t)4096 * 1024 * 2);
  half_t* w2T    = (half_t*)alloc((size_t)1024 * 4096 * 2);
  half_t* o16    = h16;   // reuse: h dead after QKV GEMM
  half_t* r216   = q16;   // reuse: q dead after attention

  // 1) weights -> transposed f16
  convert_transpose_kernel<<<(1024 * 3072 + 255) / 256, 256, 0, stream>>>(Wqkv, wqkvT, 1024, 3072);
  convert_transpose_kernel<<<(1024 * 1024 + 255) / 256, 256, 0, stream>>>(Wo, woT, 1024, 1024);
  convert_transpose_kernel<<<(1024 * 4096 + 255) / 256, 256, 0, stream>>>(W1, w1T, 1024, 4096);
  convert_transpose_kernel<<<(4096 * 1024 + 255) / 256, 256, 0, stream>>>(W2, w2T, 4096, 1024);

  // 2) LN1
  ln_kernel<<<BT, 256, 0, stream>>>(x, g1, be1, h16, nullptr);

  // 3) QKV projection + head scatter
  gemm_qkv_kernel<<<dim3(3072 / 64, BT / 128), 256, 0, stream>>>(h16, wqkvT, bqkv, q16, k16, v16);

  // 4) attention (B*H*(T/128) = 2048 blocks)
  attn_kernel<<<2048, 256, 0, stream>>>(q16, k16, v16, o16);

  // 5) output projection + residual -> r (f32)
  gemm_kernel<<<dim3(1024 / 64, BT / 128), 256, 0, stream>>>(
      o16, woT, bo, x, r32, nullptr, (int)BT, 1024, 1024, 0);

  // 6) LN2 -> r2 (f32 for residual + f16 for GEMM)
  ln_kernel<<<BT, 256, 0, stream>>>(r32, g2, be2, r216, r232);

  // 7) FFN up + ReLU
  gemm_kernel<<<dim3(4096 / 64, BT / 128), 256, 0, stream>>>(
      r216, w1T, b1, nullptr, nullptr, f116, (int)BT, 4096, 1024, 1);

  // 8) FFN down + bias + r2 residual -> out
  gemm_kernel<<<dim3(1024 / 64, BT / 128), 256, 0, stream>>>(
      f116, w2T, b2, r232, (float*)d_out, nullptr, (int)BT, 1024, 4096, 0);

  (void)in_sizes; (void)n_in; (void)out_size; (void)ws_size;
}
